// Nucleus1TransformerMoEBlock_30167850287418
// MI455X (gfx1250) — matrix operand order repaired and checked by reading, not compiled, not run on hardware
//
#include <hip/hip_runtime.h>

// ---------------------------------------------------------------------------
// Problem constants (from reference)
// ---------------------------------------------------------------------------
#define EMBED_D   768
#define QKV_N     2304          // 3*D
#define INTER_D   3072          // 4*D
#define N_EXPERTS 8
#define N_HEADS   12
#define HEAD_D    64
#define BATCH     2
#define SEQ       2048
#define T_TOK     (BATCH*SEQ)   // 4096

typedef __attribute__((ext_vector_type(16))) __bf16 v16bf;
typedef __attribute__((ext_vector_type(8)))  float  v8f;
typedef __attribute__((ext_vector_type(8)))  unsigned short ushort8;

__device__ __forceinline__ unsigned short f2bf(float f) {
    unsigned int u = __float_as_uint(f);
    u += 0x7FFFu + ((u >> 16) & 1u);          // round-to-nearest-even
    return (unsigned short)(u >> 16);
}
__device__ __forceinline__ float bf2f(unsigned short h) {
    return __uint_as_float(((unsigned int)h) << 16);
}
__device__ __forceinline__ v8f vzero() {
    v8f z = {0.f,0.f,0.f,0.f,0.f,0.f,0.f,0.f};
    return z;
}

// Fragment = two contiguous 8-half runs (callers use contiguous k = 16*(lane/16) + s).
__device__ __forceinline__ v16bf load_frag(const unsigned short* p0,
                                           const unsigned short* p1) {
    union { v16bf v; ushort8 h[2]; } u;
    u.h[0] = *(const ushort8*)p0;
    u.h[1] = *(const ushort8*)p1;
    return u.v;
}

// CDNA5 async global->LDS copy (16B per lane), tracked on ASYNCcnt.
__device__ __forceinline__ void async_load_b128(unsigned lds_off, const void* gaddr) {
    asm volatile("global_load_async_to_lds_b128 %0, %1, off"
                 :: "v"(lds_off), "v"(gaddr) : "memory");
}
__device__ __forceinline__ void wait_async0() {
    asm volatile("s_wait_asynccnt 0" ::: "memory");
}

// xor-shuffle within rows of 16 lanes via DPP16 ROW_XMASK (no DS traffic)
template<int M>
__device__ __forceinline__ float xor16f(float v) {
    return __int_as_float(__builtin_amdgcn_update_dpp(
        0, __float_as_int(v), 0x160 | M, 0xf, 0xf, true));
}

// ---------------------------------------------------------------------------
// Utility kernels
// ---------------------------------------------------------------------------
__global__ void zf32_kernel(float* p, size_t n) {
    size_t i = (size_t)blockIdx.x * blockDim.x + threadIdx.x;
    size_t st = (size_t)gridDim.x * blockDim.x;
    for (; i < n; i += st) p[i] = 0.f;
}

__global__ void cvt_bf16_kernel(const float* __restrict__ in,
                                unsigned short* __restrict__ out, size_t n) {
    size_t i = (size_t)blockIdx.x * blockDim.x + threadIdx.x;
    size_t st = (size_t)gridDim.x * blockDim.x;
    for (; i < n; i += st) out[i] = f2bf(in[i]);
}

// ---------------------------------------------------------------------------
// LayerNorm: one 256-thread block per token row -> bf16 output
// ---------------------------------------------------------------------------
__global__ __launch_bounds__(256)
void ln_kernel(const float* __restrict__ x, const float* __restrict__ g,
               const float* __restrict__ b, unsigned short* __restrict__ h) {
    __shared__ float red[256];
    int tok = blockIdx.x;
    int tid = threadIdx.x;
    const float* row = x + (size_t)tok * EMBED_D;

    float s = 0.f;
    for (int i = tid; i < EMBED_D; i += 256) s += row[i];
    red[tid] = s; __syncthreads();
    for (int o = 128; o > 0; o >>= 1) { if (tid < o) red[tid] += red[tid+o]; __syncthreads(); }
    float mu = red[0] * (1.f / EMBED_D);
    __syncthreads();

    float v = 0.f;
    for (int i = tid; i < EMBED_D; i += 256) { float d = row[i]-mu; v += d*d; }
    red[tid] = v; __syncthreads();
    for (int o = 128; o > 0; o >>= 1) { if (tid < o) red[tid] += red[tid+o]; __syncthreads(); }
    float rs = rsqrtf(red[0] * (1.f / EMBED_D) + 1e-5f);

    unsigned short* orow = h + (size_t)tok * EMBED_D;
    for (int i = tid; i < EMBED_D; i += 256)
        orow[i] = f2bf((row[i]-mu)*rs*g[i] + b[i]);
}

// ---------------------------------------------------------------------------
// Generic WMMA bf16 GEMM: C[M,N] = act(A[M,K] @ B[K,N] + bias) (+epilogues)
// Tile: 64x64 per workgroup (8 waves), BK = 32.
// A tile staged row-major via async global->LDS; B tile staged *transposed*
// ([n][k], row stride 40 halves) so B fragments are contiguous b128 loads.
// ---------------------------------------------------------------------------
#define EPI_BF16    0
#define EPI_RESID   1
#define EPI_GELU    2
#define EPI_SCATTER 3

#define BT_STRIDE 40   // 32 k-halves + 8 pad; 80B rows keep 16B alignment

template<int EPI>
__global__ __launch_bounds__(256)
void gemm_wmma_kernel(const unsigned short* __restrict__ A, int lda,
                      const unsigned short* __restrict__ Bm, int ldb,
                      const float* __restrict__ bias,
                      const float* __restrict__ resid,
                      unsigned short* __restrict__ Cbf,
                      float* __restrict__ Cf,
                      int M, int N, int K, int ldc,
                      const int*  __restrict__ gidx_a,   // gather A rows
                      const int*  __restrict__ sidx,     // scatter rows
                      const float* __restrict__ swgt,    // scatter weights
                      const int*  __restrict__ cnt)      // optional row count
{
    int Meff = cnt ? cnt[0] : M;
    int m0 = blockIdx.y * 64;
    if (m0 >= Meff) return;
    int n0 = blockIdx.x * 64;

    __shared__ __attribute__((aligned(16))) unsigned short As[64*32];
    __shared__ __attribute__((aligned(16))) unsigned short BsT[64*BT_STRIDE];

    int tid  = threadIdx.x;
    int lane = tid & 31;
    int wave = tid >> 5;
    int wm   = wave & 3;
    int wn0  = wave >> 2;

    v8f acc0 = vzero(), acc1 = vzero();

    int am  = lane & 15;
    int kh8 = (lane >> 4) * 16;
    int bn  = lane & 15;
    int bkh = (lane >> 4) * 16;

    for (int k0 = 0; k0 < K; k0 += 32) {
        __syncthreads();
        {   // A tile 64x32, row-major copy-through: async global->LDS
            int r = tid >> 2, c = (tid & 3) * 8;
            int grow = m0 + r;
            int arow = gidx_a ? ((grow < Meff) ? gidx_a[grow] : 0)
                              : ((grow < M) ? grow : 0);
            async_load_b128((unsigned)(uintptr_t)&As[r*32 + c],
                            &A[(size_t)arow * lda + k0 + c]);
        }
        {   // B tile 32x64 -> transposed LDS [n][k]
            int r = tid >> 3, c = (tid & 7) * 8;
            ushort8 bv = *(const ushort8*)&Bm[(size_t)(k0 + r) * ldb + n0 + c];
            #pragma unroll
            for (int j = 0; j < 8; ++j) BsT[(c + j) * BT_STRIDE + r] = bv[j];
        }
        if (k0 + 32 < K) {  // hint next B tile (global_prefetch_b8)
            __builtin_prefetch(&Bm[(size_t)(k0 + 32 + (tid>>3)) * ldb + n0], 0, 0);
        }
        wait_async0();
        __syncthreads();

        const unsigned short* ap = &As[(wm*16 + am)*32];
        v16bf av = load_frag(&ap[kh8], &ap[kh8 + 8]);

        const unsigned short* bp0 = &BsT[(size_t)(wn0*16 + bn)*BT_STRIDE + bkh];
        const unsigned short* bp1 = &BsT[(size_t)((wn0+2)*16 + bn)*BT_STRIDE + bkh];
        v16bf bv0 = load_frag(bp0, bp0 + 8);
        v16bf bv1 = load_frag(bp1, bp1 + 8);

        acc0 = __builtin_amdgcn_wmma_f32_16x16x32_bf16(false, av, false, bv0,
                                                       (short)0, acc0, false, false);
        acc1 = __builtin_amdgcn_wmma_f32_16x16x32_bf16(false, av, false, bv1,
                                                       (short)0, acc1, false, false);
    }

    // Epilogue. C layout: lane n = lane&15; VGPR r -> row m = r + 8*(lane>>4).
    int cn = lane & 15;
    int mh = (lane >> 4) * 8;
    #pragma unroll
    for (int r = 0; r < 8; ++r) {
        int grow = m0 + wm*16 + mh + r;
        if (grow >= Meff) continue;
        #pragma unroll
        for (int t = 0; t < 2; ++t) {
            int nt  = wn0 + 2*t;
            int col = n0 + nt*16 + cn;
            float val = (t ? acc1[r] : acc0[r]) + (bias ? bias[col] : 0.f);
            if (EPI == EPI_BF16) {
                Cbf[(size_t)grow*ldc + col] = f2bf(val);
            } else if (EPI == EPI_RESID) {
                val += resid[(size_t)grow*ldc + col];
                Cbf[(size_t)grow*ldc + col] = f2bf(val);
            } else if (EPI == EPI_GELU) {
                val = 0.5f * val * (1.f + erff(val * 0.70710678118f));
                Cbf[(size_t)grow*ldc + col] = f2bf(val);
            } else { // EPI_SCATTER
                int   tokr = sidx[grow];
                float w    = swgt[grow];
                atomicAdd(&Cf[(size_t)tokr*ldc + col], w * val);
            }
        }
    }
}

// ---------------------------------------------------------------------------
// Flash attention: grid (S/128, B*H). 8 waves; each wave owns 16 query rows.
// Q/K staged async (row-major); V staged transposed; P through per-wave LDS.
// Softmax row reductions via DPP16 ROW_XMASK (VALU, no DS).
// ---------------------------------------------------------------------------
__global__ __launch_bounds__(256)
void attn_kernel(const unsigned short* __restrict__ qkv,
                 unsigned short* __restrict__ ctx)
{
    int bh = blockIdx.y;
    int b  = bh / N_HEADS;
    int h  = bh % N_HEADS;
    int q0 = blockIdx.x * 128;
    int tid = threadIdx.x, lane = tid & 31, wave = tid >> 5;

    const unsigned short* qbase = qkv + ((size_t)b*SEQ)*QKV_N + h*HEAD_D;
    const unsigned short* kbase = qbase + EMBED_D;
    const unsigned short* vbase = qbase + 2*EMBED_D;

    __shared__ __attribute__((aligned(16))) unsigned short Qs[128*64];
    __shared__ __attribute__((aligned(16))) unsigned short Ks[32*64];
    __shared__ __attribute__((aligned(16))) unsigned short VsT[64*BT_STRIDE];
    __shared__ __attribute__((aligned(16))) unsigned short Ps[8][16*32];

    // stage Q block 128x64 via async copies
    for (int base = 0; base < 128*64; base += 256*8) {
        int e = base + tid*8;
        int row = e >> 6, col = e & 63;
        async_load_b128((unsigned)(uintptr_t)&Qs[e],
                        &qbase[(size_t)(q0+row)*QKV_N + col]);
    }
    wait_async0();
    __syncthreads();

    int am  = lane & 15;
    int kh8 = (lane >> 4) * 16;
    int bn  = lane & 15;
    int bkh = (lane >> 4) * 16;

    const unsigned short* qp = &Qs[(wave*16 + am)*64];
    v16bf qf0 = load_frag(&qp[kh8],      &qp[kh8 + 8]);    // dh 0..31
    v16bf qf1 = load_frag(&qp[32 + kh8], &qp[40 + kh8]);   // dh 32..63

    v8f acc[4] = { vzero(), vzero(), vzero(), vzero() };
    float mrun[8], lrun[8];
    #pragma unroll
    for (int r = 0; r < 8; ++r) { mrun[r] = -1e30f; lrun[r] = 0.f; }

    const float scale = 0.125f;  // 1/sqrt(64)

    for (int kb = 0; kb < SEQ; kb += 32) {
        __syncthreads();
        {   // stage K (async, row-major) and V (transposed [dh][key])
            int e = tid*8;
            int row = e >> 6, col = e & 63;
            async_load_b128((unsigned)(uintptr_t)&Ks[e],
                            &kbase[(size_t)(kb+row)*QKV_N + col]);
            ushort8 vv = *(const ushort8*)&vbase[(size_t)(kb+row)*QKV_N + col];
            #pragma unroll
            for (int j = 0; j < 8; ++j) VsT[(col + j) * BT_STRIDE + row] = vv[j];
        }
        wait_async0();
        __syncthreads();

        // scores: two 16x16 tiles (keys 0..15 / 16..31), K-dim 64 (2 steps)
        v8f s0 = vzero(), s1 = vzero();
        {
            const unsigned short* kr0 = &Ks[(size_t)bn*64];
            const unsigned short* kr1 = &Ks[(size_t)(16 + bn)*64];
            v16bf b00 = load_frag(&kr0[bkh],      &kr0[bkh + 8]);
            v16bf b01 = load_frag(&kr0[32 + bkh], &kr0[40 + bkh]);
            v16bf b10 = load_frag(&kr1[bkh],      &kr1[bkh + 8]);
            v16bf b11 = load_frag(&kr1[32 + bkh], &kr1[40 + bkh]);
            s0 = __builtin_amdgcn_wmma_f32_16x16x32_bf16(false, qf0, false, b00, (short)0, s0, false, false);
            s0 = __builtin_amdgcn_wmma_f32_16x16x32_bf16(false, qf1, false, b01, (short)0, s0, false, false);
            s1 = __builtin_amdgcn_wmma_f32_16x16x32_bf16(false, qf0, false, b10, (short)0, s1, false, false);
            s1 = __builtin_amdgcn_wmma_f32_16x16x32_bf16(false, qf1, false, b11, (short)0, s1, false, false);
        }

        // online softmax per row (reduce across 16 lanes via DPP row_xmask)
        #pragma unroll
        for (int r = 0; r < 8; ++r) {
            float a0 = s0[r]*scale, a1 = s1[r]*scale;
            float mx = fmaxf(a0, a1);
            mx = fmaxf(mx, xor16f<1>(mx));
            mx = fmaxf(mx, xor16f<2>(mx));
            mx = fmaxf(mx, xor16f<4>(mx));
            mx = fmaxf(mx, xor16f<8>(mx));
            float mnew = fmaxf(mrun[r], mx);
            float corr = __expf(mrun[r] - mnew);
            float p0 = __expf(a0 - mnew);
            float p1 = __expf(a1 - mnew);
            float ps = p0 + p1;
            ps += xor16f<1>(ps);
            ps += xor16f<2>(ps);
            ps += xor16f<4>(ps);
            ps += xor16f<8>(ps);
            lrun[r] = lrun[r]*corr + ps;
            mrun[r] = mnew;
            acc[0][r] *= corr; acc[1][r] *= corr;
            acc[2][r] *= corr; acc[3][r] *= corr;
            // stash P tile (row m = r + 8*half, cols n and 16+n)
            unsigned short* pr = &Ps[wave][(r + ((lane>>4)<<3))*32];
            pr[lane & 15]        = f2bf(p0);
            pr[16 + (lane & 15)] = f2bf(p1);
        }
        // P as A-frag (16x32 over the 32 keys); DS ops in-order per wave
        const unsigned short* pp = &Ps[wave][am*32];
        v16bf pf = load_frag(&pp[kh8], &pp[kh8 + 8]);

        // PV: 4 output n-tiles over Dh=64; V frags contiguous from VsT
        #pragma unroll
        for (int ot = 0; ot < 4; ++ot) {
            const unsigned short* vp = &VsT[(size_t)(ot*16 + bn)*BT_STRIDE + bkh];
            v16bf vf = load_frag(vp, vp + 8);
            acc[ot] = __builtin_amdgcn_wmma_f32_16x16x32_bf16(false, pf, false, vf,
                                                              (short)0, acc[ot], false, false);
        }
    }

    // normalize + store ctx (bf16) [T, D] at column h*64
    int cn = lane & 15;
    int mh = (lane >> 4) * 8;
    #pragma unroll
    for (int r = 0; r < 8; ++r) {
        int q = q0 + wave*16 + mh + r;
        float inv = 1.f / lrun[r];
        size_t orow = ((size_t)b*SEQ + q)*EMBED_D + h*HEAD_D;
        #pragma unroll
        for (int ot = 0; ot < 4; ++ot)
            ctx[orow + ot*16 + cn] = f2bf(acc[ot][r] * inv);
    }
}

// ---------------------------------------------------------------------------
// Router: logits/softmax/top-2, build per-expert gather lists + LB stats
// ---------------------------------------------------------------------------
__global__ __launch_bounds__(256)
void router_kernel(const unsigned short* __restrict__ t,
                   const float* __restrict__ w_router,
                   int* __restrict__ counts, int* __restrict__ idx,
                   float* __restrict__ wgt, float* __restrict__ sum_p)
{
    int tok = blockIdx.x * blockDim.x + threadIdx.x;
    if (tok >= T_TOK) return;
    const unsigned short* row = t + (size_t)tok * EMBED_D;
    float lg[N_EXPERTS];
    #pragma unroll
    for (int e = 0; e < N_EXPERTS; ++e) lg[e] = 0.f;
    for (int k = 0; k < EMBED_D; ++k) {
        float a = bf2f(row[k]);
        #pragma unroll
        for (int e = 0; e < N_EXPERTS; ++e)
            lg[e] += a * w_router[(size_t)k*N_EXPERTS + e];
    }
    float mx = lg[0];
    #pragma unroll
    for (int e = 1; e < N_EXPERTS; ++e) mx = fmaxf(mx, lg[e]);
    float s = 0.f, p[N_EXPERTS];
    #pragma unroll
    for (int e = 0; e < N_EXPERTS; ++e) { p[e] = __expf(lg[e]-mx); s += p[e]; }
    float invs = 1.f / s;
    #pragma unroll
    for (int e = 0; e < N_EXPERTS; ++e) p[e] *= invs;

    int i0 = 0;
    #pragma unroll
    for (int e = 1; e < N_EXPERTS; ++e) if (p[e] > p[i0]) i0 = e;
    int i1 = (i0 == 0) ? 1 : 0;
    #pragma unroll
    for (int e = 0; e < N_EXPERTS; ++e) if (e != i0 && p[e] > p[i1]) i1 = e;

    float inv = 1.f / (p[i0] + p[i1]);
    float c0 = p[i0]*inv, c1 = p[i1]*inv;

    int p0 = atomicAdd(&counts[i0], 1);
    idx[i0*T_TOK + p0] = tok; wgt[i0*T_TOK + p0] = c0;
    int p1 = atomicAdd(&counts[i1], 1);
    idx[i1*T_TOK + p1] = tok; wgt[i1*T_TOK + p1] = c1;
    #pragma unroll
    for (int e = 0; e < N_EXPERTS; ++e) atomicAdd(&sum_p[e], p[e]);
}

__global__ void lb_kernel(const int* __restrict__ counts,
                          const float* __restrict__ sum_p,
                          float* __restrict__ out_scalar)
{
    if (threadIdx.x == 0 && blockIdx.x == 0) {
        float acc = 0.f;
        for (int e = 0; e < N_EXPERTS; ++e)
            acc += ((float)counts[e] / (float)T_TOK) * (sum_p[e] / (float)T_TOK);
        *out_scalar = 0.01f * (float)N_EXPERTS * acc;
    }
}

// ---------------------------------------------------------------------------
// Host launcher
// ---------------------------------------------------------------------------
extern "C" void kernel_launch(void* const* d_in, const int* in_sizes, int n_in,
                              void* d_out, int out_size, void* d_ws, size_t ws_size,
                              hipStream_t stream) {
    const float* x        = (const float*)d_in[0];
    const float* ln_g     = (const float*)d_in[1];
    const float* ln_b     = (const float*)d_in[2];
    const float* w_qkv    = (const float*)d_in[3];
    const float* b_qkv    = (const float*)d_in[4];
    const float* w_o      = (const float*)d_in[5];
    const float* b_o      = (const float*)d_in[6];
    const float* w_router = (const float*)d_in[7];
    const float* w1       = (const float*)d_in[8];
    const float* b1       = (const float*)d_in[9];
    const float* w2       = (const float*)d_in[10];
    const float* b2       = (const float*)d_in[11];
    float* out = (float*)d_out;                 // T*D floats + 1 (lb loss)

    // workspace carve-up
    char* wp = (char*)d_ws;
    auto alloc = [&](size_t bytes) -> void* {
        void* p = wp; wp += (bytes + 255) & ~(size_t)255; return p;
    };
    unsigned short* h_bf    = (unsigned short*)alloc((size_t)T_TOK*EMBED_D*2);
    unsigned short* qkv_bf  = (unsigned short*)alloc((size_t)T_TOK*QKV_N*2);
    unsigned short* ctx_bf  = (unsigned short*)alloc((size_t)T_TOK*EMBED_D*2);
    unsigned short* t_bf    = (unsigned short*)alloc((size_t)T_TOK*EMBED_D*2);
    unsigned short* act_bf  = (unsigned short*)alloc((size_t)T_TOK*INTER_D*2);
    unsigned short* wqkv_bf = (unsigned short*)alloc((size_t)EMBED_D*QKV_N*2);
    unsigned short* wo_bf   = (unsigned short*)alloc((size_t)EMBED_D*EMBED_D*2);
    unsigned short* w1_bf   = (unsigned short*)alloc((size_t)N_EXPERTS*EMBED_D*INTER_D*2);
    unsigned short* w2_bf   = (unsigned short*)alloc((size_t)N_EXPERTS*INTER_D*EMBED_D*2);
    int*   counts = (int*)alloc(N_EXPERTS*4);
    float* sum_p  = (float*)alloc(N_EXPERTS*4);
    int*   gidx   = (int*)alloc((size_t)N_EXPERTS*T_TOK*4);
    float* gwgt   = (float*)alloc((size_t)N_EXPERTS*T_TOK*4);

    // 1) convert weights f32 -> bf16
    cvt_bf16_kernel<<<1024, 256, 0, stream>>>(w_qkv, wqkv_bf, (size_t)EMBED_D*QKV_N);
    cvt_bf16_kernel<<<1024, 256, 0, stream>>>(w_o,   wo_bf,   (size_t)EMBED_D*EMBED_D);
    cvt_bf16_kernel<<<4096, 256, 0, stream>>>(w1,    w1_bf,   (size_t)N_EXPERTS*EMBED_D*INTER_D);
    cvt_bf16_kernel<<<4096, 256, 0, stream>>>(w2,    w2_bf,   (size_t)N_EXPERTS*INTER_D*EMBED_D);

    // 2) zero accumulators / routing buffers / output
    zf32_kernel<<<1024, 256, 0, stream>>>(out, (size_t)T_TOK*EMBED_D + 1);
    zf32_kernel<<<1,    256, 0, stream>>>((float*)counts, N_EXPERTS);
    zf32_kernel<<<1,    256, 0, stream>>>(sum_p, N_EXPERTS);
    zf32_kernel<<<256,  256, 0, stream>>>((float*)gidx, (size_t)N_EXPERTS*T_TOK);
    zf32_kernel<<<256,  256, 0, stream>>>(gwgt,         (size_t)N_EXPERTS*T_TOK);

    // 3) LayerNorm -> h (bf16)
    ln_kernel<<<T_TOK, 256, 0, stream>>>(x, ln_g, ln_b, h_bf);

    // 4) QKV GEMM: [T,768] x [768,2304] -> bf16
    gemm_wmma_kernel<EPI_BF16><<<dim3(QKV_N/64, T_TOK/64), 256, 0, stream>>>(
        h_bf, EMBED_D, wqkv_bf, QKV_N, b_qkv, nullptr,
        qkv_bf, nullptr, T_TOK, QKV_N, EMBED_D, QKV_N,
        nullptr, nullptr, nullptr, nullptr);

    // 5) attention (flash, WMMA)
    attn_kernel<<<dim3(SEQ/128, BATCH*N_HEADS), 256, 0, stream>>>(qkv_bf, ctx_bf);

    // 6) out-proj + residual: t = ctx @ w_o + b_o + x  -> bf16
    gemm_wmma_kernel<EPI_RESID><<<dim3(EMBED_D/64, T_TOK/64), 256, 0, stream>>>(
        ctx_bf, EMBED_D, wo_bf, EMBED_D, b_o, x,
        t_bf, nullptr, T_TOK, EMBED_D, EMBED_D, EMBED_D,
        nullptr, nullptr, nullptr, nullptr);

    // 7) router: softmax/top-2/gather lists + LB stats
    router_kernel<<<T_TOK/256, 256, 0, stream>>>(t_bf, w_router, counts, gidx, gwgt, sum_p);

    // 8) MoE experts (gathered tokens, sequential per expert)
    for (int e = 0; e < N_EXPERTS; ++e) {
        gemm_wmma_kernel<EPI_GELU><<<dim3(INTER_D/64, T_TOK/64), 256, 0, stream>>>(
            t_bf, EMBED_D, w1_bf + (size_t)e*EMBED_D*INTER_D, INTER_D,
            b1 + (size_t)e*INTER_D, nullptr,
            act_bf, nullptr, T_TOK, INTER_D, EMBED_D, INTER_D,
            gidx + (size_t)e*T_TOK, nullptr, nullptr, &counts[e]);

        gemm_wmma_kernel<EPI_SCATTER><<<dim3(EMBED_D/64, T_TOK/64), 256, 0, stream>>>(
            act_bf, INTER_D, w2_bf + (size_t)e*INTER_D*EMBED_D, EMBED_D,
            b2 + (size_t)e*EMBED_D, nullptr,
            nullptr, out, T_TOK, EMBED_D, INTER_D, EMBED_D,
            nullptr, gidx + (size_t)e*T_TOK, gwgt + (size_t)e*T_TOK, &counts[e]);
    }

    // 9) load-balance loss scalar
    lb_kernel<<<1, 32, 0, stream>>>(counts, sum_p, out + (size_t)T_TOK*EMBED_D);
}
